// Net_4105988735288
// MI455X (gfx1250) — matrix operand order repaired and checked by reading, not compiled, not run on hardware
//
#include <hip/hip_runtime.h>
#include <hip/hip_bf16.h>
#include <math.h>

#define VOCAB  32000
#define EMBED  128
#define HIDDEN 256
#define BB     16
#define SS     512

// ---------------- WMMA fragment types (gfx1250, wave32) ----------------
typedef __attribute__((ext_vector_type(16))) __bf16 v16bf;
typedef __attribute__((ext_vector_type(8)))  float  v8f;
typedef __attribute__((ext_vector_type(4)))  unsigned int v4u;

union FragAB { v16bf v; v4u q[2]; unsigned short h[16]; };
union FragC  { v8f  v; float f[8]; };

__device__ __forceinline__ unsigned short f32_to_bf16(float x) {
    union { float f; unsigned int u; } c; c.f = x;
    unsigned int u = c.u;
    u += 0x7fffu + ((u >> 16) & 1u);   // round-to-nearest-even
    return (unsigned short)(u >> 16);
}

// Native CDNA5 transcendental: v_tanh_f32 (TRANS32 op).
// ISA 7.4.3: one independent op (v_nop) required before the result is used.
__device__ __forceinline__ float fast_tanh(float x) {
    float r;
    asm volatile("v_tanh_f32 %0, %1\n\tv_nop" : "=v"(r) : "v"(x));
    return r;
}

// ---------------- Stage A: xw[s][b][h] = embed[x_idx[b,s]] @ w_i2h -----
__global__ void k_embed_i2h(const int* __restrict__ x_idx,
                            const float* __restrict__ embed,
                            const float* __restrict__ w_i2h,
                            float* __restrict__ xw) {
    __shared__ float erow[EMBED];
    const int r = blockIdx.x;            // r = b*S + s
    const int b = r / SS, s = r % SS;
    const int h = threadIdx.x;           // 0..255
    const int idx = x_idx[r];
    if (h < EMBED) erow[h] = embed[(size_t)idx * EMBED + h];
    __syncthreads();
    float acc = 0.f;
    const float* wc = w_i2h + h;         // column h, row stride HIDDEN
    #pragma unroll 8
    for (int e = 0; e < EMBED; ++e)
        acc = fmaf(erow[e], wc[(size_t)e * HIDDEN], acc);
    xw[((size_t)s * BB + b) * HIDDEN + h] = acc;   // time-major [S][B][H]
}

// -------- Transpose+convert: src f32 [K][N] -> dst bf16 [N][K] ---------
__global__ void k_transpose_bf16(const float* __restrict__ src,
                                 unsigned short* __restrict__ dst,
                                 int K, int N) {
    __shared__ float tile[32][33];
    const int kb = blockIdx.y * 32, nb = blockIdx.x * 32;
    const int tx = threadIdx.x & 31, ty = threadIdx.x >> 5;  // 256 thr: ty 0..7
    #pragma unroll
    for (int i = 0; i < 32; i += 8)
        tile[ty + i][tx] = src[(size_t)(kb + ty + i) * N + (nb + tx)];
    __syncthreads();
    #pragma unroll
    for (int i = 0; i < 32; i += 8)
        dst[(size_t)(nb + ty + i) * K + (kb + tx)] = f32_to_bf16(tile[tx][ty + i]);
}

// ---------------- Stage B: sequential RNN scan (one WGP) ---------------
// 16 waves; wave w owns hidden columns [16w, 16w+16).
// w_h2h^T fragments live in registers for all 512 steps; h ping-pongs in LDS.
__launch_bounds__(512, 1)
__global__ void k_rnn_scan(const float* __restrict__ xw,
                           const float* __restrict__ h0,
                           const unsigned short* __restrict__ wT_h2h,
                           const float* __restrict__ b_h2h,
                           unsigned short* __restrict__ hs) {
    __shared__ __align__(16) unsigned short hbuf[2][BB][HIDDEN];
    const int tid  = threadIdx.x;
    const int wave = tid >> 5;
    const int lane = tid & 31;
    const int half = lane >> 4;
    const int l16  = lane & 15;
    const int col  = wave * 16 + l16;

    // Preload B fragments (w_h2h^T row = our output column), 8 k-tiles.
    FragAB bfrag[8];
    {
        const unsigned short* row = wT_h2h + (size_t)col * HIDDEN;
        #pragma unroll
        for (int kt = 0; kt < 8; ++kt) {
            const v4u* p = (const v4u*)(row + kt * 32 + half * 8);
            bfrag[kt].q[0] = p[0];
            bfrag[kt].q[1] = p[2];
        }
    }
    const float bias = b_h2h[col];

    for (int i = tid; i < BB * HIDDEN; i += 512)
        ((unsigned short*)hbuf[0])[i] = f32_to_bf16(h0[i]);
    __syncthreads();

    for (int t = 0; t < SS; ++t) {
        const int cur = t & 1, nxt = cur ^ 1;

        // C init = xw_t + bias, in the 16x16 f32 C/D layout. xw is read-once: NT.
        FragC acc;
        #pragma unroll
        for (int i = 0; i < 8; ++i) {
            const int b = i + half * 8;
            acc.f[i] = __builtin_nontemporal_load(
                           &xw[((size_t)t * BB + b) * HIDDEN + col]) + bias;
        }
        // acc += h @ w_h2h  (8 chained WMMAs over K=256)
        #pragma unroll
        for (int kt = 0; kt < 8; ++kt) {
            FragAB a;
            const unsigned short* arow = &hbuf[cur][l16][0];
            a.q[0] = *(const v4u*)(arow + kt * 32 + half * 8);
            a.q[1] = *(const v4u*)(arow + kt * 32 + 16 + half * 8);
            acc.v = __builtin_amdgcn_wmma_f32_16x16x32_bf16(
                        false, a.v, false, bfrag[kt].v,
                        (short)0, acc.v, false, false);
        }
        // h_new = tanh(acc) via native v_tanh_f32, write to the other buffer.
        #pragma unroll
        for (int i = 0; i < 8; ++i) {
            const int b = i + half * 8;
            hbuf[nxt][b][col] = f32_to_bf16(fast_tanh(acc.f[i]));
        }
        __syncthreads();
        // Persist h_t to global hs[b*S + t][:], coalesced from LDS.
        for (int i = tid; i < BB * HIDDEN; i += 512) {
            const int b = i >> 8, hc = i & 255;
            hs[((size_t)b * SS + t) * HIDDEN + hc] = hbuf[nxt][b][hc];
        }
    }
}

// ------- Stage C: logits[8192,32000] = hs @ w_h2o + b (bf16 WMMA) ------
// 8 waves/block; each wave computes a 32x64 tile (2 M-tiles x 4 N-tiles).
// hs (4MB) and w_h2o^T (16MB) stay hot in the 192MB L2; the 1GB output is
// streamed with non-temporal stores so it does not evict them.
__launch_bounds__(256)
__global__ void k_out_gemm(const unsigned short* __restrict__ hs,
                           const unsigned short* __restrict__ wT,
                           const float* __restrict__ b_h2o,
                           float* __restrict__ out) {
    const int tid  = threadIdx.x;
    const int wave = tid >> 5;
    const int lane = tid & 31;
    const int half = lane >> 4;
    const int l16  = lane & 15;

    const long job = (long)blockIdx.x * 8 + wave;   // 128000 wave-jobs
    const int NJ = VOCAB / 64;                       // 500
    const int mj = (int)(job / NJ), nj = (int)(job % NJ);
    const int rowbase = mj * 32, colbase = nj * 64;

    FragC acc[2][4];
    #pragma unroll
    for (int m = 0; m < 2; ++m)
        #pragma unroll
        for (int u = 0; u < 4; ++u)
            #pragma unroll
            for (int i = 0; i < 8; ++i) acc[m][u].f[i] = 0.f;

    const unsigned short* arow0 = hs + (size_t)(rowbase + l16) * HIDDEN;
    const unsigned short* arow1 = arow0 + (size_t)16 * HIDDEN;
    const unsigned short* brow[4];
    #pragma unroll
    for (int u = 0; u < 4; ++u) {
        brow[u] = wT + (size_t)(colbase + u * 16 + l16) * HIDDEN;
        __builtin_prefetch(brow[u], 0, 0);
    }

    #pragma unroll
    for (int kt = 0; kt < 8; ++kt) {
        FragAB a0, a1, b[4];
        a0.q[0] = *(const v4u*)(arow0 + kt * 32 + half * 8);
        a0.q[1] = *(const v4u*)(arow0 + kt * 32 + 16 + half * 8);
        a1.q[0] = *(const v4u*)(arow1 + kt * 32 + half * 8);
        a1.q[1] = *(const v4u*)(arow1 + kt * 32 + 16 + half * 8);
        #pragma unroll
        for (int u = 0; u < 4; ++u) {
            const v4u* p = (const v4u*)(brow[u] + kt * 32 + half * 8);
            b[u].q[0] = p[0];
            b[u].q[1] = p[2];
        }
        #pragma unroll
        for (int u = 0; u < 4; ++u) {
            acc[0][u].v = __builtin_amdgcn_wmma_f32_16x16x32_bf16(
                              false, a0.v, false, b[u].v, (short)0, acc[0][u].v, false, false);
            acc[1][u].v = __builtin_amdgcn_wmma_f32_16x16x32_bf16(
                              false, a1.v, false, b[u].v, (short)0, acc[1][u].v, false, false);
        }
    }

    #pragma unroll
    for (int u = 0; u < 4; ++u) {
        const int c = colbase + u * 16 + l16;
        const float bias = b_h2o[c];
        #pragma unroll
        for (int m = 0; m < 2; ++m) {
            #pragma unroll
            for (int i = 0; i < 8; ++i) {
                const int r = rowbase + m * 16 + i + half * 8;
                __builtin_nontemporal_store(acc[m][u].f[i] + bias,
                                            &out[(size_t)r * VOCAB + c]);
            }
        }
    }
}

// ----------------------------- launcher --------------------------------
extern "C" void kernel_launch(void* const* d_in, const int* in_sizes, int n_in,
                              void* d_out, int out_size, void* d_ws, size_t ws_size,
                              hipStream_t stream) {
    const int*   x_idx = (const int*)  d_in[0];
    const float* h_0   = (const float*)d_in[1];
    const float* embed = (const float*)d_in[2];
    const float* w_i2h = (const float*)d_in[3];
    const float* w_h2h = (const float*)d_in[4];
    const float* b_h2h = (const float*)d_in[5];
    const float* w_h2o = (const float*)d_in[6];
    const float* b_h2o = (const float*)d_in[7];
    float* out = (float*)d_out;

    // Workspace carve (all 16B aligned): xw f32 (8MB), w_h2h^T bf16 (128KB),
    // hs bf16 (4MB), w_h2o^T bf16 (16MB)  => ~28.2 MB total.
    char* ws = (char*)d_ws;
    float* xw = (float*)ws;                      ws += (size_t)SS * BB * HIDDEN * sizeof(float);
    unsigned short* wT_h2h = (unsigned short*)ws; ws += (size_t)HIDDEN * HIDDEN * sizeof(unsigned short);
    unsigned short* hs     = (unsigned short*)ws; ws += (size_t)BB * SS * HIDDEN * sizeof(unsigned short);
    unsigned short* wT_h2o = (unsigned short*)ws;

    // A: embedding gather + input projection (time-major xw)
    k_embed_i2h<<<BB * SS, HIDDEN, 0, stream>>>(x_idx, embed, w_i2h, xw);
    // Pre-transpose weights into [N][K] bf16 for WMMA B-fragment loads
    k_transpose_bf16<<<dim3(HIDDEN / 32, HIDDEN / 32), 256, 0, stream>>>(w_h2h, wT_h2h, HIDDEN, HIDDEN);
    k_transpose_bf16<<<dim3(VOCAB / 32,  HIDDEN / 32), 256, 0, stream>>>(w_h2o, wT_h2o, HIDDEN, VOCAB);
    // B: sequential recurrence on a single WGP (16 waves, WMMA + LDS ping-pong)
    k_rnn_scan<<<1, 512, 0, stream>>>(xw, h_0, wT_h2h, b_h2h, hs);
    // C: big output GEMM, store-bandwidth bound
    k_out_gemm<<<(BB * SS / 32) * (VOCAB / 64) / 8, 256, 0, stream>>>(hs, wT_h2o, b_h2o, out);
}
